// BertSelfAttention_2851858284557
// MI455X (gfx1250) — matrix operand order repaired and checked by reading, not compiled, not run on hardware
//
#include <hip/hip_runtime.h>
#include <hip/hip_bf16.h>

typedef _Float16 h16;
typedef __attribute__((ext_vector_type(8)))  _Float16 v8h;
typedef __attribute__((ext_vector_type(16))) _Float16 v16h;
typedef __attribute__((ext_vector_type(8)))  float    v8f;

#define B_   2
#define S_   2048
#define D_   768
#define H_   12
#define W_   64
#define M_   (B_*S_)     // 4096
#define NQKV (3*D_)      // 2304

static __device__ __forceinline__ v16h join16(v8h lo, v8h hi) {
  return __builtin_shufflevector(lo, hi, 0,1,2,3,4,5,6,7,8,9,10,11,12,13,14,15);
}

// Async 16B copy global->LDS (GVS mode: saddr base + 32-bit byte offset).
// Tracked with ASYNCcnt; LDS byte offset = low 32 bits of generic pointer.
static __device__ __forceinline__ void async_cp16(const h16* base, unsigned gbyte,
                                                  const h16* ldsp) {
  unsigned laddr = (unsigned)(unsigned long long)ldsp;
  asm volatile("global_load_async_to_lds_b128 %0, %1, %2"
               :: "v"(laddr), "v"(gbyte), "s"(base) : "memory");
}

// ---------------- Kernel A: fp32 -> fp16 packing ----------------
__global__ void pack_f16_kernel(const float* __restrict__ x,
                                const float* __restrict__ Wq,
                                const float* __restrict__ Wk,
                                const float* __restrict__ Wv,
                                h16* __restrict__ xh, h16* __restrict__ wh) {
  const int XN = M_ * D_;       // 3145728
  const int WN = NQKV * D_;     // 1769472
  int stride = gridDim.x * blockDim.x;
  for (int i = blockIdx.x * blockDim.x + threadIdx.x; i < XN; i += stride)
    xh[i] = (h16)x[i];
  for (int i = blockIdx.x * blockDim.x + threadIdx.x; i < WN; i += stride) {
    int n = i / D_, k = i - n * D_;
    int sel = n / D_;                 // 0=q,1=k,2=v (n in [0,2304))
    int row = n - sel * D_;
    const float* Wm = (sel == 0) ? Wq : ((sel == 1) ? Wk : Wv);
    wh[i] = (h16)Wm[row * D_ + k];
  }
}

// ---------------- Kernel B: QKV projection GEMM (WMMA f16, async pipelined) --
#define BK  32
#define LDA 40   // 32 + 8 halves pad; stride 80B (16B aligned, conflict-free)
#define NKT (D_/BK)   // 24
__global__ __launch_bounds__(256)
void qkv_gemm_kernel(const h16* __restrict__ xh, const h16* __restrict__ wh,
                     const float* __restrict__ bq, const float* __restrict__ bk,
                     const float* __restrict__ bv,
                     h16* __restrict__ qout, h16* __restrict__ kout,
                     h16* __restrict__ vout) {
  __shared__ __align__(16) h16 As[2][128 * LDA];
  __shared__ __align__(16) h16 Bs[2][128 * LDA];
  const int tid = threadIdx.x;
  const int lane = tid & 31, wave = tid >> 5;
  const int wm = wave & 3, wn = wave >> 2;   // wave tile: 32 rows x 64 cols
  const int m0 = blockIdx.y * 128;
  const int n0 = blockIdx.x * 128;
  const int lr = lane & 15, lh = lane >> 4;

  v8f acc[2][4] = {};

  // issue async copies for k-step `kt` into buffer `buf` (4 ops / thread)
  auto issue_tile = [&](int kt, int buf) {
    const int kbase = kt * BK;
    #pragma unroll
    for (int i = 0; i < 2; ++i) {
      int c = tid + i * 256;          // 512 x 16B chunks each for A and B
      int r = c >> 2, cp = c & 3;
      unsigned ga = (unsigned)(((m0 + r) * D_ + kbase + cp * 8) * 2);
      async_cp16(xh, ga, &As[buf][r * LDA + cp * 8]);
      unsigned gb = (unsigned)(((n0 + r) * D_ + kbase + cp * 8) * 2);
      async_cp16(wh, gb, &Bs[buf][r * LDA + cp * 8]);
    }
  };

  issue_tile(0, 0);
  for (int kt = 0; kt < NKT; ++kt) {
    const int cur = kt & 1;
    if (kt + 1 < NKT) {
      issue_tile(kt + 1, cur ^ 1);                    // prefetch next tile
      asm volatile("s_wait_asynccnt 0x4" ::: "memory"); // tile kt landed
    } else {
      asm volatile("s_wait_asynccnt 0x0" ::: "memory");
    }
    __syncthreads();

    const h16* Ab = As[cur];
    const h16* Bb = Bs[cur];
    v16h afr[2];
    #pragma unroll
    for (int fm = 0; fm < 2; ++fm) {
      const h16* p = Ab + (wm * 32 + fm * 16 + lr) * LDA + lh * 8;
      afr[fm] = join16(*(const v8h*)p, *(const v8h*)(p + 16));
    }
    v16h bfr[4];
    #pragma unroll
    for (int fn = 0; fn < 4; ++fn) {
      const h16* p = Bb + (wn * 64 + fn * 16 + lr) * LDA + lh * 8;
      bfr[fn] = join16(*(const v8h*)p, *(const v8h*)(p + 16));
    }
    #pragma unroll
    for (int fn = 0; fn < 4; ++fn)
      #pragma unroll
      for (int fm = 0; fm < 2; ++fm)
        acc[fm][fn] = __builtin_amdgcn_wmma_f32_16x16x32_f16(
            false, afr[fm], false, bfr[fn], (short)0, acc[fm][fn], false, false);
    __syncthreads();   // safe to overwrite this buffer two k-steps later
  }

  // Epilogue: bias + scatter f16. q,k: (B,H,S,W); v: (B,H,W,S) pre-transposed.
  #pragma unroll
  for (int fn = 0; fn < 4; ++fn) {
    int n = n0 + wn * 64 + fn * 16 + lr;
    int sel = n / D_, nn = n - sel * D_;
    int hh = nn >> 6, wc = nn & 63;
    float bias = (sel == 0) ? bq[nn] : ((sel == 1) ? bk[nn] : bv[nn]);
    h16* outp = (sel == 0) ? qout : ((sel == 1) ? kout : vout);
    #pragma unroll
    for (int fm = 0; fm < 2; ++fm) {
      #pragma unroll
      for (int v = 0; v < 8; ++v) {
        int m = m0 + wm * 32 + fm * 16 + v + lh * 8;
        int bb = m >> 11, ss = m & 2047;
        float val = acc[fm][fn][v] + bias;
        size_t idx = (sel == 2)
            ? (((size_t)(bb * H_ + hh)) * W_ + wc) * S_ + ss
            : (((size_t)(bb * H_ + hh)) * S_ + ss) * W_ + wc;
        outp[idx] = (h16)val;
      }
    }
  }
}

// ---------------- Kernel C: flash attention (WMMA f16, async double-buffer) -
#define LDK 72   // 64 + 8 halves pad; stride 144B (16B aligned)
#define NJT (S_/64)   // 32
__global__ __launch_bounds__(256)
void attn_kernel(const h16* __restrict__ qb, const h16* __restrict__ kb,
                 const h16* __restrict__ vtb, const int* __restrict__ mask,
                 float* __restrict__ out) {
  __shared__ __align__(16) h16 Kt[2][64 * LDK];   // [t][w]
  __shared__ __align__(16) h16 Vt[2][64 * LDK];   // [w][t] (from (B,H,W,S))
  __shared__ __align__(16) h16 Ps[128 * LDK];     // per-wave P tiles [m][t]
  const int tid = threadIdx.x, lane = tid & 31, wave = tid >> 5;
  const int lr = lane & 15, lh = lane >> 4;
  const int bh = blockIdx.y;
  const int bI = bh / H_, hI = bh % H_;
  const h16* qp = qb  + (size_t)bh * S_ * W_;   // (s,w) rows
  const h16* kp = kb  + (size_t)bh * S_ * W_;   // (t,w) rows
  const h16* vp = vtb + (size_t)bh * W_ * S_;   // (w,t) rows
  const int s0 = blockIdx.x * 128 + wave * 16;

  v16h aq[2];
  #pragma unroll
  for (int kk = 0; kk < 2; ++kk) {
    const h16* p = qp + (size_t)(s0 + lr) * W_ + kk * 32 + lh * 8;
    aq[kk] = join16(*(const v8h*)p, *(const v8h*)(p + 16));
  }

  float mrun[8], lrun[8];
  #pragma unroll
  for (int v = 0; v < 8; ++v) { mrun[v] = -1e30f; lrun[v] = 0.f; }
  v8f octx[4] = {};

  // issue async K/V tile copies for key-tile jt into buffer buf (4 ops/thread)
  auto issue_tile = [&](int jt, int buf) {
    const int t0 = jt * 64;
    #pragma unroll
    for (int i = 0; i < 2; ++i) {
      int c = tid + i * 256;          // 512 x 16B chunks each
      int rr = c >> 3, part = c & 7;
      unsigned gk = (unsigned)((((t0 + rr) * W_) + part * 8) * 2);
      async_cp16(kp, gk, &Kt[buf][rr * LDK + part * 8]);
      unsigned gv = (unsigned)(((rr * S_) + t0 + part * 8) * 2);
      async_cp16(vp, gv, &Vt[buf][rr * LDK + part * 8]);
    }
  };

  issue_tile(0, 0);
  for (int jt = 0; jt < NJT; ++jt) {
    const int cur = jt & 1;
    const int t0 = jt * 64;
    if (jt + 1 < NJT) {
      issue_tile(jt + 1, cur ^ 1);
      asm volatile("s_wait_asynccnt 0x4" ::: "memory");
    } else {
      asm volatile("s_wait_asynccnt 0x0" ::: "memory");
    }
    __syncthreads();

    const h16* Kb = Kt[cur];
    const h16* Vb = Vt[cur];

    // S = Q K^T
    v8f sacc[4] = {};
    #pragma unroll
    for (int fn = 0; fn < 4; ++fn) {
      #pragma unroll
      for (int kk = 0; kk < 2; ++kk) {
        const h16* p = Kb + (fn * 16 + lr) * LDK + kk * 32 + lh * 8;
        v16h bf = join16(*(const v8h*)p, *(const v8h*)(p + 16));
        sacc[fn] = __builtin_amdgcn_wmma_f32_16x16x32_f16(
            false, aq[kk], false, bf, (short)0, sacc[fn], false, false);
      }
    }
    int mok[4];
    #pragma unroll
    for (int fn = 0; fn < 4; ++fn)
      mok[fn] = mask[bI * S_ + t0 + fn * 16 + lr];

    // online softmax per row (rows live across 16-lane halves)
    #pragma unroll
    for (int v = 0; v < 8; ++v) {
      float sv[4];
      #pragma unroll
      for (int fn = 0; fn < 4; ++fn)
        sv[fn] = mok[fn] ? sacc[fn][v] * 0.125f : -1e20f;
      float rmax = fmaxf(fmaxf(sv[0], sv[1]), fmaxf(sv[2], sv[3]));
      #pragma unroll
      for (int off = 1; off < 16; off <<= 1)
        rmax = fmaxf(rmax, __shfl_xor(rmax, off, 32));
      float mnew = fmaxf(mrun[v], rmax);
      float alpha = __expf(mrun[v] - mnew);
      float rsum = 0.f;
      int prow = wave * 16 + v + lh * 8;
      #pragma unroll
      for (int fn = 0; fn < 4; ++fn) {
        float pv = __expf(sv[fn] - mnew);
        rsum += pv;
        Ps[prow * LDK + fn * 16 + lr] = (h16)pv;
      }
      #pragma unroll
      for (int off = 1; off < 16; off <<= 1)
        rsum += __shfl_xor(rsum, off, 32);
      lrun[v] = lrun[v] * alpha + rsum;
      mrun[v] = mnew;
      #pragma unroll
      for (int fn = 0; fn < 4; ++fn)
        octx[fn][v] *= alpha;
    }

    // O += P V  (per-wave Ps region; same-wave LDS ordering)
    #pragma unroll
    for (int kk = 0; kk < 2; ++kk) {
      const h16* p = Ps + (wave * 16 + lr) * LDK + kk * 32 + lh * 8;
      v16h ap = join16(*(const v8h*)p, *(const v8h*)(p + 16));
      #pragma unroll
      for (int fn = 0; fn < 4; ++fn) {
        const h16* pb = Vb + (fn * 16 + lr) * LDK + kk * 32 + lh * 8;
        v16h bf = join16(*(const v8h*)pb, *(const v8h*)(pb + 16));
        octx[fn] = __builtin_amdgcn_wmma_f32_16x16x32_f16(
            false, ap, false, bf, (short)0, octx[fn], false, false);
      }
    }
    __syncthreads();   // all waves done with this K/V buffer before refill
  }

  // normalize + write hidden (b,s,h*64+w) fp32
  #pragma unroll
  for (int v = 0; v < 8; ++v) {
    float inv = 1.f / lrun[v];
    int ss = s0 + v + lh * 8;
    #pragma unroll
    for (int fn = 0; fn < 4; ++fn) {
      int col = hI * W_ + fn * 16 + lr;
      out[((size_t)(bI * S_ + ss)) * D_ + col] = octx[fn][v] * inv;
    }
  }
}

extern "C" void kernel_launch(void* const* d_in, const int* in_sizes, int n_in,
                              void* d_out, int out_size, void* d_ws, size_t ws_size,
                              hipStream_t stream) {
  const float* x  = (const float*)d_in[0];
  const float* Wq = (const float*)d_in[1];
  const float* bq = (const float*)d_in[2];
  const float* Wk = (const float*)d_in[3];
  const float* bk = (const float*)d_in[4];
  const float* Wv = (const float*)d_in[5];
  const float* bv = (const float*)d_in[6];
  const int* mask = (const int*)d_in[7];
  float* out = (float*)d_out;

  char* ws = (char*)d_ws;
  h16* xh = (h16*)(ws);                    // 4096*768*2  = 6,291,456 B
  h16* wh = (h16*)(ws + 6291456);          // 2304*768*2  = 3,538,944 B
  h16* qb = (h16*)(ws + 9830400);          // 2*12*2048*64*2 = 6,291,456 B
  h16* kb = (h16*)(ws + 16121856);
  h16* vb = (h16*)(ws + 22413312);         // v stored (B,H,W,S); total 28,704,768 B

  pack_f16_kernel<<<1024, 256, 0, stream>>>(x, Wq, Wk, Wv, xh, wh);
  qkv_gemm_kernel<<<dim3(NQKV / 128, M_ / 128), 256, 0, stream>>>(
      xh, wh, bq, bk, bv, qb, kb, vb);
  attn_kernel<<<dim3(S_ / 128, B_ * H_), 256, 0, stream>>>(
      qb, kb, vb, mask, out);
}
